// RIGNO_44006234915055
// MI455X (gfx1250) — matrix operand order repaired and checked by reading, not compiled, not run on hardware
//
#include <hip/hip_runtime.h>
#include <hip/hip_bf16.h>
#include <stdint.h>

// ---------------------------------------------------------------------------
// RIGNO GraphCast-style GNN for MI455X (gfx1250, wave32, WMMA).
// All MLPs (K in {32pad,128,256,384} -> 128 -> 128) run on
// v_wmma_f32_16x16x32_bf16: weights/activations bf16, accum f32.
// Working set (~170MB) fits the 192MB L2 -> compute-bound -> matrix cores.
// Invariants exploited: every MLP row count is a multiple of 128, every
// concat segment width is a multiple of 32 (embeds pre-padded to 32).
// ---------------------------------------------------------------------------

typedef __attribute__((ext_vector_type(16))) __bf16 v16bf;
typedef __attribute__((ext_vector_type(8)))  float  v8f;

union Frag16 {
  v16bf v;
  __bf16 h[16];
  unsigned short u[16];
  uint32_t d[8];
  uint4 q[2];
};

#define HS_STRIDE 136   // halfwords; 68 dwords/row -> bank-conflict-free frag reads

__device__ __forceinline__ unsigned short f2bf(float f) {
  union { float f; uint32_t u; } c; c.f = f;
  uint32_t r = c.u + 0x7FFFu + ((c.u >> 16) & 1u);   // round-to-nearest-even
  return (unsigned short)(r >> 16);
}
__device__ __forceinline__ uint32_t pack_bf2(float lo, float hi) {
  return (uint32_t)f2bf(lo) | ((uint32_t)f2bf(hi) << 16);
}

// Pack an f32 weight matrix W[K][N] (N<=128, zero-padded) into WMMA B-fragment
// layout: dword index = ((ks*8 + ntile)*32 + lane)*8 + v, each dword = bf16 pair
// (k, k+1) for column n = ntile*16 + lane%16, k = ks*32 + 2*v + 16*(lane/16).
__global__ __launch_bounds__(256) void pack_w_kernel(
    const float* __restrict__ W, int K, int N, int Kpad, uint32_t* __restrict__ dst) {
  int t = blockIdx.x * 256 + threadIdx.x;
  int total = Kpad * 64;
  if (t >= total) return;
  int v    = t & 7;
  int lane = (t >> 3) & 31;
  int nt   = (t >> 8) & 7;
  int ks   = t >> 11;
  int half = lane >> 4, l15 = lane & 15;
  int n = nt * 16 + l15;
  int k = ks * 32 + 2 * v + 16 * half;
  float w0 = (k     < K && n < N) ? W[(size_t)k       * N + n] : 0.f;
  float w1 = ((k+1) < K && n < N) ? W[(size_t)(k + 1) * N + n] : 0.f;
  dst[t] = pack_bf2(w0, w1);
}

// dst[row][0:32] = concat(a[row][0:wa], b[row][0:wb], zeros)
__global__ __launch_bounds__(256) void concat_pad_kernel(
    const float* __restrict__ a, int wa, const float* __restrict__ b, int wb,
    float* __restrict__ dst, int M) {
  int t = blockIdx.x * 256 + threadIdx.x;
  int row = t >> 5, c = t & 31;
  if (row >= M) return;
  float v = 0.f;
  if (c < wa) v = a[(size_t)row * wa + c];
  else if (b && c - wa < wb) v = b[(size_t)row * wb + (c - wa)];
  dst[(size_t)row * 32 + c] = v;
}

__global__ void zero_kernel(float* __restrict__ p, int n) {
  int i = blockIdx.x * 256 + threadIdx.x;
  if (i < n) p[i] = 0.f;
}

// segment-sum: agg[recv[e], :] += src[e, :]
__global__ void scatter_add_kernel(const float* __restrict__ src, const int* __restrict__ recv,
                                   float* __restrict__ agg, int E) {
  int t = blockIdx.x * 256 + threadIdx.x;
  int edge = t >> 7, col = t & 127;
  if (edge < E) atomicAdd(agg + (size_t)recv[edge] * 128 + col, src[(size_t)edge * 128 + col]);
}

// Fused: out[row] = (res?res[row]:0) + LN(swish(X@W1+b1)@W2+b2)*g+bt
// X = concat of up to 3 (optionally gathered) f32 segments; all widths and M
// multiples of 32 / 128 respectively (guaranteed by launcher).
__global__ __launch_bounds__(256)
void mlp_wmma_kernel(
    const float* __restrict__ x0, const int* __restrict__ i0, int w0,
    const float* __restrict__ x1, const int* __restrict__ i1, int w1,
    const float* __restrict__ x2, const int* __restrict__ i2, int w2,
    const uint32_t* __restrict__ Wp1, const float* __restrict__ b1,
    const uint32_t* __restrict__ Wp2, const float* __restrict__ b2,
    const float* __restrict__ g, const float* __restrict__ bt,
    const float* __restrict__ res,
    float* __restrict__ out, int dstride, int dout, int M)
{
  __shared__ unsigned short Hs[128 * HS_STRIDE];   // ~34KB hidden tile (bf16)

  const int lane = threadIdx.x & 31;
  const int wave = threadIdx.x >> 5;
  const int l15  = lane & 15;
  const int half = lane >> 4;
  const int tile = blockIdx.x * 128;
  const int kstot = (w0 + w1 + w2) >> 5;
  (void)M;

  // Row this lane feeds into the A operand of its wave's 16-row strip.
  const int rowA = tile + wave * 16 + l15;
  const float* p0 = x0 + (size_t)(i0 ? i0[rowA] : rowA) * w0;
  const float* p1 = x1 ? x1 + (size_t)(i1 ? i1[rowA] : rowA) * w1 : nullptr;
  const float* p2 = x2 ? x2 + (size_t)(i2 ? i2[rowA] : rowA) * w2 : nullptr;

  const v8f vzero = {0.f,0.f,0.f,0.f,0.f,0.f,0.f,0.f};
  v8f acc[8];
  #pragma unroll
  for (int nt = 0; nt < 8; ++nt) acc[nt] = vzero;

  // ---- GEMM1: [16 x K] @ [K x 128]; one branch-free loop per segment ------
  int ksg = 0;   // global K-step (indexes weight fragments)
  auto gemm1_seg = [&](const float* __restrict__ ps0, int wseg) {
    const int nks = wseg >> 5;
    for (int ks2 = 0; ks2 < nks; ++ks2, ++ksg) {
      const float* ps = ps0 + ks2 * 32;
      Frag16 a;
      #pragma unroll
      for (int v = 0; v < 8; ++v) {
        // A layout (contiguous, matches packed B): k = 2v + 16*half
        const int koff = 2 * v + 16 * half;
        const float2 xv = *(const float2*)(ps + koff);   // 8B-aligned pair
        a.h[2 * v]     = (__bf16)xv.x;                   // hw RNE convert
        a.h[2 * v + 1] = (__bf16)xv.y;
      }
      const uint4* wq = (const uint4*)Wp1 + ((size_t)ksg * 256 + lane) * 2;
      if (ksg + 1 < kstot)
        __builtin_prefetch((const uint4*)Wp1 + ((size_t)(ksg + 1) * 256 + lane) * 2, 0, 1);
      Frag16 b[8];
      #pragma unroll
      for (int nt = 0; nt < 8; ++nt) {       // clause of b128 loads
        b[nt].q[0] = wq[nt * 64];
        b[nt].q[1] = wq[nt * 64 + 1];
      }
      #pragma unroll
      for (int nt = 0; nt < 8; ++nt)         // back-to-back WMMAs
        acc[nt] = __builtin_amdgcn_wmma_f32_16x16x32_bf16(
            false, a.v, false, b[nt].v, (short)0, acc[nt], false, false);
    }
  };
  gemm1_seg(p0, w0);
  if (p1) gemm1_seg(p1, w1);
  if (p2) gemm1_seg(p2, w2);

  // ---- epilogue1: +b1, swish, C-layout -> LDS row-major bf16 ---------------
  #pragma unroll
  for (int nt = 0; nt < 8; ++nt) {
    const int n = nt * 16 + l15;
    const float bb = b1[n];
    #pragma unroll
    for (int r = 0; r < 8; ++r) {
      float x = acc[nt][r] + bb;
      float h = x / (1.f + __expf(-x));                 // swish
      Hs[(wave * 16 + r + 8 * half) * HS_STRIDE + n] = f2bf(h);
    }
  }
  __syncthreads();

  // ---- GEMM2: [16 x 128] @ [128 x 128] ------------------------------------
  v8f acc2[8];
  #pragma unroll
  for (int nt = 0; nt < 8; ++nt) acc2[nt] = vzero;
  const unsigned short* hrow = &Hs[(wave * 16 + l15) * HS_STRIDE];
  #pragma unroll
  for (int ks = 0; ks < 4; ++ks) {
    Frag16 a;
    #pragma unroll
    for (int v = 0; v < 8; ++v) {
      const int k = ks * 32 + 2 * v + 16 * half;
      a.d[v] = *(const uint32_t*)(hrow + k);            // dword = bf16 pair
    }
    const uint4* wq = (const uint4*)Wp2 + ((size_t)ks * 256 + lane) * 2;
    Frag16 b[8];
    #pragma unroll
    for (int nt = 0; nt < 8; ++nt) {
      b[nt].q[0] = wq[nt * 64];
      b[nt].q[1] = wq[nt * 64 + 1];
    }
    #pragma unroll
    for (int nt = 0; nt < 8; ++nt)
      acc2[nt] = __builtin_amdgcn_wmma_f32_16x16x32_bf16(
          false, a.v, false, b[nt].v, (short)0, acc2[nt], false, false);
  }

  // ---- epilogue2: +b2, LayerNorm (16-lane shfl reduction), residual, store -
  if (g) {
    #pragma unroll
    for (int nt = 0; nt < 8; ++nt) {
      const float bb = b2[nt * 16 + l15];
      #pragma unroll
      for (int r = 0; r < 8; ++r) acc2[nt][r] += bb;
    }
    #pragma unroll
    for (int r = 0; r < 8; ++r) {
      float s = 0.f, q = 0.f;
      #pragma unroll
      for (int nt = 0; nt < 8; ++nt) { float v = acc2[nt][r]; s += v; q += v * v; }
      #pragma unroll
      for (int m = 1; m < 16; m <<= 1) {
        s += __shfl_xor(s, m, 16);
        q += __shfl_xor(q, m, 16);
      }
      const float mean = s * (1.f / 128.f);
      const float var  = q * (1.f / 128.f) - mean * mean;
      const float sc   = rsqrtf(var + 1e-5f);
      const int row = tile + wave * 16 + r + 8 * half;
      #pragma unroll
      for (int nt = 0; nt < 8; ++nt) {
        const int n = nt * 16 + l15;
        float val = (acc2[nt][r] - mean) * sc * g[n] + bt[n];
        if (res) val += res[(size_t)row * 128 + n];
        out[(size_t)row * dstride + n] = val;
      }
    }
  } else {
    #pragma unroll
    for (int r = 0; r < 8; ++r) {
      const int row = tile + wave * 16 + r + 8 * half;
      #pragma unroll
      for (int nt = 0; nt < 8; ++nt) {
        const int n = nt * 16 + l15;
        if (n < dout) {
          float val = acc2[nt][r] + b2[n];
          if (res) val += res[(size_t)row * 128 + n];
          out[(size_t)row * dstride + n] = val;
        }
      }
    }
  }
}

// ---------------------------------------------------------------------------
extern "C" void kernel_launch(void* const* d_in, const int* in_sizes, int n_in,
                              void* d_out, int out_size, void* d_ws, size_t ws_size,
                              hipStream_t stream) {
  (void)in_sizes; (void)n_in; (void)out_size; (void)ws_size;
  const int Np = 32768, Nr = 16384, Ep = 131072, Er = 98304, Ed = 131072, S = 18;

  const float* u      = (const float*)d_in[0];
  const float* pnsf   = (const float*)d_in[1];
  const float* rnsf   = (const float*)d_in[2];
  const float* p2r_ef = (const float*)d_in[3];
  const float* r2r_ef = (const float*)d_in[4];
  const float* r2p_ef = (const float*)d_in[5];
  const int* p2r_s = (const int*)d_in[6];
  const int* p2r_r = (const int*)d_in[7];
  const int* r2r_s = (const int*)d_in[8];
  const int* r2r_r = (const int*)d_in[9];
  const int* r2p_s = (const int*)d_in[10];
  const int* r2p_r = (const int*)d_in[11];

  // params flattened with sorted dict keys (JAX pytree order):
  // decE,decN,ee_p2r,ee_r2p,ee_r2r,encE,encN,out,pe,pro{E,N},re
  // each MLP: W1,W2,b1,b2[,bt,g]
  struct P { const float *W1, *W2, *b1, *b2, *bt, *g; };
  auto getP = [&](int base, bool ln) {
    P p;
    p.W1 = (const float*)d_in[base + 0];
    p.W2 = (const float*)d_in[base + 1];
    p.b1 = (const float*)d_in[base + 2];
    p.b2 = (const float*)d_in[base + 3];
    p.bt = ln ? (const float*)d_in[base + 4] : nullptr;
    p.g  = ln ? (const float*)d_in[base + 5] : nullptr;
    return p;
  };
  P decE   = getP(12, true);
  P decN   = getP(18, true);
  P ee_p2r = getP(24, true);
  P ee_r2p = getP(30, true);
  P ee_r2r = getP(36, true);
  P encE   = getP(42, true);
  P encN   = getP(48, true);
  P outp   = getP(54, false);   // W1,W2,b1,b2 only
  P pe     = getP(58, true);
  P proE   = getP(64, true);    // stacked [18,...]
  P proN   = getP(70, true);    // stacked [18,...]
  P re     = getP(76, true);

  // ---- workspace carve ----------------------------------------------------
  uintptr_t base = (uintptr_t)d_ws;
  size_t off = 0;
  auto carve = [&](size_t bytes) -> void* {
    void* p = (void*)(base + off);
    off += (bytes + 255) & ~(size_t)255;
    return p;
  };
  float* pn   = (float*)carve((size_t)Np * 128 * 4);
  float* rn   = (float*)carve((size_t)Nr * 128 * 4);
  float* e    = (float*)carve((size_t)Ep * 128 * 4);   // reused as ed in decoder
  float* er   = (float*)carve((size_t)Er * 128 * 4);
  float* agg  = (float*)carve((size_t)Np * 128 * 4);
  float* xpad = (float*)carve((size_t)Ep * 32 * 4);    // padded embed inputs
  auto carveW = [&](int Kpad) { return (uint32_t*)carve((size_t)Kpad * 256); };

  uint32_t *pk_pe1   = carveW(32),  *pk_pe2   = carveW(128);
  uint32_t *pk_re1   = carveW(32),  *pk_re2   = carveW(128);
  uint32_t *pk_eeP1  = carveW(32),  *pk_eeP2  = carveW(128);
  uint32_t *pk_eeR1  = carveW(32),  *pk_eeR2  = carveW(128);
  uint32_t *pk_eeD1  = carveW(32),  *pk_eeD2  = carveW(128);
  uint32_t *pk_encE1 = carveW(384), *pk_encE2 = carveW(128);
  uint32_t *pk_encN1 = carveW(256), *pk_encN2 = carveW(128);
  uint32_t *pk_decE1 = carveW(384), *pk_decE2 = carveW(128);
  uint32_t *pk_decN1 = carveW(256), *pk_decN2 = carveW(128);
  uint32_t *pk_out1  = carveW(128), *pk_out2  = carveW(128);
  uint32_t *pk_proE1 = (uint32_t*)carve((size_t)S * 384 * 256);
  uint32_t *pk_proE2 = (uint32_t*)carve((size_t)S * 128 * 256);
  uint32_t *pk_proN1 = (uint32_t*)carve((size_t)S * 256 * 256);
  uint32_t *pk_proN2 = (uint32_t*)carve((size_t)S * 128 * 256);

  // ---- helpers ------------------------------------------------------------
  auto pack = [&](const float* W, int K, int N, int Kpad, uint32_t* dst) {
    int total = Kpad * 64;
    pack_w_kernel<<<(total + 255) / 256, 256, 0, stream>>>(W, K, N, Kpad, dst);
  };
  auto mlp = [&](const float* x0, const int* i0, int w0,
                 const float* x1, const int* i1, int w1,
                 const float* x2, const int* i2, int w2,
                 const uint32_t* Wp1, const float* b1,
                 const uint32_t* Wp2, const float* b2,
                 const float* g, const float* bt,
                 const float* res, float* out, int dstride, int dout, int M) {
    mlp_wmma_kernel<<<M / 128, 256, 0, stream>>>(
        x0, i0, w0, x1, i1, w1, x2, i2, w2,
        Wp1, b1, Wp2, b2, g, bt, res, out, dstride, dout, M);
  };
  auto cpad = [&](const float* a, int wa, const float* b, int wb, int M) {
    concat_pad_kernel<<<(M * 32 + 255) / 256, 256, 0, stream>>>(a, wa, b, wb, xpad, M);
  };
  auto zero = [&](float* p, int n) {
    zero_kernel<<<(n + 255) / 256, 256, 0, stream>>>(p, n);
  };
  auto scatter = [&](const float* src, const int* recv, float* dst, int E) {
    scatter_add_kernel<<<(E * 128 + 255) / 256, 256, 0, stream>>>(src, recv, dst, E);
  };

  // ---- pack all weights to bf16 WMMA fragments (deterministic each call) --
  pack(pe.W1, 5, 128, 32, pk_pe1);          pack(pe.W2, 128, 128, 128, pk_pe2);
  pack(re.W1, 2, 128, 32, pk_re1);          pack(re.W2, 128, 128, 128, pk_re2);
  pack(ee_p2r.W1, 3, 128, 32, pk_eeP1);     pack(ee_p2r.W2, 128, 128, 128, pk_eeP2);
  pack(ee_r2r.W1, 3, 128, 32, pk_eeR1);     pack(ee_r2r.W2, 128, 128, 128, pk_eeR2);
  pack(ee_r2p.W1, 3, 128, 32, pk_eeD1);     pack(ee_r2p.W2, 128, 128, 128, pk_eeD2);
  pack(encE.W1, 384, 128, 384, pk_encE1);   pack(encE.W2, 128, 128, 128, pk_encE2);
  pack(encN.W1, 256, 128, 256, pk_encN1);   pack(encN.W2, 128, 128, 128, pk_encN2);
  pack(decE.W1, 384, 128, 384, pk_decE1);   pack(decE.W2, 128, 128, 128, pk_decE2);
  pack(decN.W1, 256, 128, 256, pk_decN1);   pack(decN.W2, 128, 128, 128, pk_decN2);
  pack(outp.W1, 128, 128, 128, pk_out1);    pack(outp.W2, 128, 3, 128, pk_out2);
  for (int s = 0; s < S; ++s) {
    pack(proE.W1 + (size_t)s * 384 * 128, 384, 128, 384, pk_proE1 + (size_t)s * 384 * 64);
    pack(proE.W2 + (size_t)s * 128 * 128, 128, 128, 128, pk_proE2 + (size_t)s * 128 * 64);
    pack(proN.W1 + (size_t)s * 256 * 128, 256, 128, 256, pk_proN1 + (size_t)s * 256 * 64);
    pack(proN.W2 + (size_t)s * 128 * 128, 128, 128, 128, pk_proN2 + (size_t)s * 128 * 64);
  }

  // ---- node/edge embeddings (inputs pre-padded to one 32-wide segment) ----
  cpad(u, 3, pnsf, 2, Np);
  mlp(xpad, nullptr, 32, nullptr, nullptr, 0, nullptr, nullptr, 0,
      pk_pe1, pe.b1, pk_pe2, pe.b2, pe.g, pe.bt, nullptr, pn, 128, 128, Np);
  cpad(rnsf, 2, nullptr, 0, Nr);
  mlp(xpad, nullptr, 32, nullptr, nullptr, 0, nullptr, nullptr, 0,
      pk_re1, re.b1, pk_re2, re.b2, re.g, re.bt, nullptr, rn, 128, 128, Nr);

  // ---- encoder: pmesh -> rmesh -------------------------------------------
  cpad(p2r_ef, 3, nullptr, 0, Ep);
  mlp(xpad, nullptr, 32, nullptr, nullptr, 0, nullptr, nullptr, 0,
      pk_eeP1, ee_p2r.b1, pk_eeP2, ee_p2r.b2, ee_p2r.g, ee_p2r.bt, nullptr, e, 128, 128, Ep);
  mlp(e, nullptr, 128, pn, p2r_s, 128, rn, p2r_r, 128,
      pk_encE1, encE.b1, pk_encE2, encE.b2, encE.g, encE.bt, e, e, 128, 128, Ep);
  zero(agg, Nr * 128);
  scatter(e, p2r_r, agg, Ep);
  mlp(rn, nullptr, 128, agg, nullptr, 128, nullptr, nullptr, 0,
      pk_encN1, encN.b1, pk_encN2, encN.b2, encN.g, encN.bt, rn, rn, 128, 128, Nr);

  // ---- processor: 18 interaction-network steps ---------------------------
  cpad(r2r_ef, 3, nullptr, 0, Er);
  mlp(xpad, nullptr, 32, nullptr, nullptr, 0, nullptr, nullptr, 0,
      pk_eeR1, ee_r2r.b1, pk_eeR2, ee_r2r.b2, ee_r2r.g, ee_r2r.bt, nullptr, er, 128, 128, Er);
  for (int s = 0; s < S; ++s) {
    mlp(er, nullptr, 128, rn, r2r_s, 128, rn, r2r_r, 128,
        pk_proE1 + (size_t)s * 384 * 64, proE.b1 + (size_t)s * 128,
        pk_proE2 + (size_t)s * 128 * 64, proE.b2 + (size_t)s * 128,
        proE.g + (size_t)s * 128, proE.bt + (size_t)s * 128,
        er, er, 128, 128, Er);
    zero(agg, Nr * 128);
    scatter(er, r2r_r, agg, Er);
    mlp(rn, nullptr, 128, agg, nullptr, 128, nullptr, nullptr, 0,
        pk_proN1 + (size_t)s * 256 * 64, proN.b1 + (size_t)s * 128,
        pk_proN2 + (size_t)s * 128 * 64, proN.b2 + (size_t)s * 128,
        proN.g + (size_t)s * 128, proN.bt + (size_t)s * 128,
        rn, rn, 128, 128, Nr);
  }

  // ---- decoder: rmesh -> pmesh -------------------------------------------
  float* ed = e;  // reuse encoder edge buffer
  cpad(r2p_ef, 3, nullptr, 0, Ed);
  mlp(xpad, nullptr, 32, nullptr, nullptr, 0, nullptr, nullptr, 0,
      pk_eeD1, ee_r2p.b1, pk_eeD2, ee_r2p.b2, ee_r2p.g, ee_r2p.bt, nullptr, ed, 128, 128, Ed);
  mlp(ed, nullptr, 128, rn, r2p_s, 128, pn, r2p_r, 128,
      pk_decE1, decE.b1, pk_decE2, decE.b2, decE.g, decE.bt, ed, ed, 128, 128, Ed);
  zero(agg, Np * 128);
  scatter(ed, r2p_r, agg, Ed);
  mlp(pn, nullptr, 128, agg, nullptr, 128, nullptr, nullptr, 0,
      pk_decN1, decN.b1, pk_decN2, decN.b2, decN.g, decN.bt, pn, pn, 128, 128, Np);

  // ---- output head: 128 -> 128 -> 3, no LN -------------------------------
  mlp(pn, nullptr, 128, nullptr, nullptr, 0, nullptr, nullptr, 0,
      pk_out1, outp.b1, pk_out2, outp.b2, nullptr, nullptr, nullptr,
      (float*)d_out, 3, 3, Np);
}
